// WaveNet_86526411145922
// MI455X (gfx1250) — matrix operand order repaired and checked by reading, not compiled, not run on hardware
//
#include <hip/hip_runtime.h>
#include <hip/hip_bf16.h>

#define BB    4
#define TT    8000
#define TMEL  32
#define CCH   80
#define CCP   96      // cond channels padded to 96 (zeros in 80..95)
#define RR    64
#define SS    256
#define OO    256
#define LL    16
#define UPW   1024
#define UPS   256
#define KCAT  224     // 64 (x[t-d]) + 64 (x[t]) + 96 (cond, padded)
#define MGATE 128
#define MRS   320     // 64 res rows + 256 skip rows

typedef __bf16 bf16;
typedef bf16  bf16x8 __attribute__((ext_vector_type(8)));
typedef bf16  v16bf  __attribute__((ext_vector_type(16)));
typedef float v8f    __attribute__((ext_vector_type(8)));

// ---------------------------------------------------------------- helpers
__device__ __forceinline__ v8f wmma_bf16(v16bf a, v16bf b, v8f c) {
    return __builtin_amdgcn_wmma_f32_16x16x32_bf16(false, a, false, b,
                                                   (short)0, c, false, false);
}

// A-operand (row-major M x K, 16-bit): lane (m,h) holds K[32k+16h .. +15]
// (two 16B chunks)
__device__ __forceinline__ v16bf load_a(const bf16* row, int k, int hh) {
    bf16x8 lo = *(const bf16x8*)(row + 32 * k + 16 * hh);
    bf16x8 hi = *(const bf16x8*)(row + 32 * k + 16 * hh + 8);
    return __builtin_shufflevector(lo, hi, 0,1,2,3,4,5,6,7,8,9,10,11,12,13,14,15);
}

// ---------------------------------------------------------------- weight prep
// Packs bf16 weights:
//  Wcat[L][128][224] = [dil_w tap0 | dil_w tap1 | cond_w rows | 0-pad]
//  RS[L][320][64]    = [res_w (0 for last layer) ; skip_w]
//  Wco/Wce[256][256] = conv_out_w / conv_end_w
//  biasin[L][128]    = dil_b + cond_b
__global__ void prep_weights(const float* __restrict__ dil_w,
                             const float* __restrict__ cond_w,
                             const float* __restrict__ res_w,
                             const float* __restrict__ skip_w,
                             const float* __restrict__ conv_out_w,
                             const float* __restrict__ conv_end_w,
                             const float* __restrict__ dil_b,
                             const float* __restrict__ cond_b,
                             bf16* __restrict__ Wcat, bf16* __restrict__ RS,
                             bf16* __restrict__ Wco, bf16* __restrict__ Wce,
                             float* __restrict__ biasin) {
    int tid = blockIdx.x * blockDim.x + threadIdx.x;
    const int NW = LL * MGATE * KCAT;   // 458752
    const int NR = LL * MRS * RR;       // 327680
    const int NC = OO * SS;             // 65536
    if (tid < NW) {
        int i = tid / (MGATE * KCAT);
        int r = tid % (MGATE * KCAT);
        int o = r / KCAT, k = r % KCAT;
        float v;
        if (k < 64)       v = dil_w[(((size_t)i * MGATE + o) * RR + k) * 2 + 0];
        else if (k < 128) v = dil_w[(((size_t)i * MGATE + o) * RR + (k - 64)) * 2 + 1];
        else if (k < 208) v = cond_w[((size_t)i * MGATE + o) * CCH + (k - 128)];
        else              v = 0.f;
        Wcat[tid] = (bf16)v;
        return;
    }
    tid -= NW;
    if (tid < NR) {
        int i = tid / (MRS * RR);
        int r = tid % (MRS * RR);
        int m = r / RR, c = r % RR;
        float v;
        if (m < RR) v = (i < LL - 1) ? res_w[((size_t)i * RR + m) * RR + c] : 0.f;
        else        v = skip_w[((size_t)i * SS + (m - RR)) * RR + c];
        RS[NW - NW + (size_t)i * MRS * RR + r] = (bf16)v;
        return;
    }
    tid -= NR;
    if (tid < NC) { Wco[tid] = (bf16)conv_out_w[tid]; return; }
    tid -= NC;
    if (tid < NC) { Wce[tid] = (bf16)conv_end_w[tid]; return; }
    tid -= NC;
    if (tid < LL * MGATE) biasin[tid] = dil_b[tid] + cond_b[tid];
}

// ---------------------------------------------------------------- cond upsample
// cond[b][t][o] = up_b[o] + sum_m sum_i feat[b][i][m] * up_w[i][o][t - m*256]
// stored time-major bf16, padded to 96 channels.
__global__ void cond_kernel(const float* __restrict__ feat,
                            const float* __restrict__ up_w,
                            const float* __restrict__ up_b,
                            bf16* __restrict__ cond) {
    int tid = blockIdx.x * blockDim.x + threadIdx.x;
    if (tid >= BB * TT * CCP) return;
    int o  = tid % CCP;
    int bt = tid / CCP;
    int t  = bt % TT, b = bt / TT;
    float acc = 0.f;
    if (o < CCH) {
        acc = up_b[o];
        int mhi = t / UPS;
        if (mhi > TMEL - 1) mhi = TMEL - 1;
        for (int m = mhi; m >= 0; --m) {
            int k = t - m * UPS;
            if (k >= UPW) break;
            const float* f = feat + (size_t)b * CCH * TMEL + m;
            const float* w = up_w + (size_t)o * UPW + k;
            for (int i = 0; i < CCH; ++i)
                acc += f[(size_t)i * TMEL] * w[(size_t)i * CCH * UPW];
        }
    }
    cond[tid] = (bf16)acc;
}

// ---------------------------------------------------------------- embedding
__global__ void embed_kernel(const int* __restrict__ fwd,
                             const float* __restrict__ embed,
                             float* __restrict__ xf, bf16* __restrict__ xb) {
    int tid = blockIdx.x * blockDim.x + threadIdx.x;
    if (tid >= BB * TT * RR) return;
    int c  = tid % RR;
    int bt = tid / RR;
    float v = embed[(size_t)fwd[bt] * RR + c];
    xf[tid] = v;
    xb[tid] = (bf16)v;
}

// ---------------------------------------------------------------- wavenet layer
// Phase 1: in_act(128 x 128cols) = Wcat @ [x[t-d]; x[t]; cond[t]]  (K=224)
// Phase 2: acts = tanh(a) * sigmoid(g)   (via LDS)
// Phase 3: [res;skip](320x64) @ acts ; residual add + skip accumulate
__global__ __launch_bounds__(256)
void layer_kernel(const bf16* __restrict__ Wcat, const float* __restrict__ biasin,
                  const bf16* __restrict__ RS, const float* __restrict__ resb,
                  const float* __restrict__ skipb,
                  const bf16* __restrict__ xin_b, const float* __restrict__ xin_f,
                  const bf16* __restrict__ cond,
                  bf16* __restrict__ xout_b, float* __restrict__ xout_f,
                  float* __restrict__ skip_acc,
                  int dil, int first, int last) {
    __shared__ bf16 lds_in[MGATE * 128];   // 32 KB: in_act+bias (bf16)
    __shared__ bf16 lds_acts[128 * RR];    // 16 KB: gated acts, time-major

    int b      = blockIdx.y;
    int tbase  = blockIdx.x * 128;
    int ntiles = (TT - tbase) / 16;
    if (ntiles > 8) ntiles = 8;
    int lane = threadIdx.x & 31, wave = threadIdx.x >> 5;
    int ml = lane & 15, hh = lane >> 4;

    // ---- Phase 1: each wave owns M-tile `wave` (rows wave*16 .. +15)
    {
        const bf16* wrow = Wcat + (size_t)(wave * 16 + ml) * KCAT;
        v16bf afr[7];
#pragma unroll
        for (int k = 0; k < 7; ++k) afr[k] = load_a(wrow, k, hh);
        float bia[8];
#pragma unroll
        for (int v = 0; v < 8; ++v) bia[v] = biasin[wave * 16 + v + 8 * hh];

        v16bf zfr = {};
        for (int nt = 0; nt < ntiles; ++nt) {
            int t  = tbase + nt * 16 + ml;
            int ts = t - dil;
            int zs = (ts < 0);
            if (zs) ts = 0;
            size_t rowx  = ((size_t)b * TT + t) * RR;
            size_t rowxd = ((size_t)b * TT + ts) * RR;
            size_t rowc  = ((size_t)b * TT + t) * CCP;
            v8f acc = {};
#pragma unroll
            for (int k = 0; k < 7; ++k) {
                int ks = 32 * k + 16 * hh;
                v16bf bfr;
                if (k < 2) {
                    bfr = *(const v16bf*)(xin_b + rowxd + ks);
                    bfr = zs ? zfr : bfr;
                } else if (k < 4) {
                    bfr = *(const v16bf*)(xin_b + rowx + (ks - 64));
                } else {
                    bfr = *(const v16bf*)(cond + rowc + (ks - 128));
                }
                acc = wmma_bf16(afr[k], bfr, acc);
            }
#pragma unroll
            for (int v = 0; v < 8; ++v)
                lds_in[(wave * 16 + v + 8 * hh) * 128 + nt * 16 + ml] =
                    (bf16)(acc[v] + bia[v]);
        }
    }
    __syncthreads();

    // ---- Phase 2: gating, store acts time-major (n, r) bf16
    {
        int ncols = ntiles * 16;
        for (int idx = threadIdx.x; idx < RR * ncols; idx += 256) {
            int r = idx & (RR - 1);
            int n = idx >> 6;
            float a = (float)lds_in[r * 128 + n];
            float g = (float)lds_in[(RR + r) * 128 + n];
            float val = tanhf(a) / (1.f + __expf(-g));
            lds_acts[n * RR + r] = (bf16)val;
        }
    }
    __syncthreads();

    // ---- Phase 3: 20 M-tiles x ntiles, wave-strided
    for (int job = wave; job < 20 * ntiles; job += 8) {
        int mt = job % 20, nt = job / 20;
        const bf16* arow = RS + (size_t)(mt * 16 + ml) * RR;
        v8f acc = {};
#pragma unroll
        for (int k = 0; k < 2; ++k) {
            v16bf afr = load_a(arow, k, hh);
            v16bf bfr = *(const v16bf*)(lds_acts + (nt * 16 + ml) * RR +
                                        32 * k + 16 * hh);
            acc = wmma_bf16(afr, bfr, acc);
        }
        int t  = tbase + nt * 16 + ml;
        int m0 = mt * 16 + 8 * hh;
        if (mt < 4) {                       // residual rows (m0 in 0..56)
            if (!last) {
                size_t base = ((size_t)b * TT + t) * RR + m0;
                float vals[8];
                float4 x0 = *(const float4*)(xin_f + base);
                float4 x1 = *(const float4*)(xin_f + base + 4);
                float xs[8] = {x0.x, x0.y, x0.z, x0.w, x1.x, x1.y, x1.z, x1.w};
#pragma unroll
                for (int v = 0; v < 8; ++v)
                    vals[v] = acc[v] + resb[m0 + v] + xs[v];
                *(float4*)(xout_f + base)     = *(float4*)(vals);
                *(float4*)(xout_f + base + 4) = *(float4*)(vals + 4);
                bf16x8 ob;
#pragma unroll
                for (int v = 0; v < 8; ++v) ob[v] = (bf16)vals[v];
                *(bf16x8*)(xout_b + base) = ob;
            }
        } else {                            // skip rows
            int sc0 = m0 - 64;
            size_t base = ((size_t)b * TT + t) * SS + sc0;
            float vals[8];
#pragma unroll
            for (int v = 0; v < 8; ++v) vals[v] = acc[v] + skipb[sc0 + v];
            if (!first) {
                float4 s0 = *(const float4*)(skip_acc + base);
                float4 s1 = *(const float4*)(skip_acc + base + 4);
                vals[0] += s0.x; vals[1] += s0.y; vals[2] += s0.z; vals[3] += s0.w;
                vals[4] += s1.x; vals[5] += s1.y; vals[6] += s1.z; vals[7] += s1.w;
            }
            *(float4*)(skip_acc + base)     = *(float4*)(vals);
            *(float4*)(skip_acc + base + 4) = *(float4*)(vals + 4);
        }
    }
}

// ---------------------------------------------------------------- conv_out (relu in, relu out)
__global__ __launch_bounds__(256)
void conv_out_kernel(const bf16* __restrict__ Wco, const float* __restrict__ skipf,
                     bf16* __restrict__ y1) {
    int b      = blockIdx.y;
    int tbase  = blockIdx.x * 128;
    int ntiles = (TT - tbase) / 16;
    if (ntiles > 8) ntiles = 8;
    int lane = threadIdx.x & 31, wave = threadIdx.x >> 5;
    int ml = lane & 15, hh = lane >> 4;

    for (int job = wave; job < 16 * ntiles; job += 8) {
        int mt = job % 16, nt = job / 16;
        const bf16* arow = Wco + (size_t)(mt * 16 + ml) * SS;
        int t = tbase + nt * 16 + ml;
        const float* brow = skipf + ((size_t)b * TT + t) * SS;
        v8f acc = {};
#pragma unroll
        for (int k = 0; k < 8; ++k) {
            v16bf afr = load_a(arow, k, hh);
            const float4* sp = (const float4*)(brow + 32 * k + 16 * hh);
            float4 f0 = sp[0], f1 = sp[1], f2 = sp[2], f3 = sp[3];
            float tmp[16];
            *(float4*)(tmp + 0) = f0; *(float4*)(tmp + 4)  = f1;
            *(float4*)(tmp + 8) = f2; *(float4*)(tmp + 12) = f3;
            v16bf bfr;
#pragma unroll
            for (int e = 0; e < 16; ++e) bfr[e] = (bf16)fmaxf(tmp[e], 0.f);
            acc = wmma_bf16(afr, bfr, acc);
        }
        int m0 = mt * 16 + 8 * hh;
        bf16x8 ov;
#pragma unroll
        for (int v = 0; v < 8; ++v) ov[v] = (bf16)fmaxf(acc[v], 0.f);
        *(bf16x8*)(y1 + ((size_t)b * TT + t) * OO + m0) = ov;
    }
}

// ---------------------------------------------------------------- conv_end + causal shift
__global__ __launch_bounds__(256)
void conv_end_kernel(const bf16* __restrict__ Wce, const bf16* __restrict__ y1,
                     float* __restrict__ out) {
    int b      = blockIdx.y;
    int tbase  = blockIdx.x * 128;
    int ntiles = (TT - tbase) / 16;
    if (ntiles > 8) ntiles = 8;
    int lane = threadIdx.x & 31, wave = threadIdx.x >> 5;
    int ml = lane & 15, hh = lane >> 4;

    for (int job = wave; job < 16 * ntiles; job += 8) {
        int mt = job % 16, nt = job / 16;
        const bf16* arow = Wce + (size_t)(mt * 16 + ml) * OO;
        int t = tbase + nt * 16 + ml;
        const bf16* brow = y1 + ((size_t)b * TT + t) * OO;
        v8f acc = {};
#pragma unroll
        for (int k = 0; k < 8; ++k) {
            v16bf afr = load_a(arow, k, hh);
            v16bf bfr = *(const v16bf*)(brow + 32 * k + 16 * hh);
            acc = wmma_bf16(afr, bfr, acc);
        }
        if (t < TT - 1) {
            int m0 = mt * 16 + 8 * hh;
#pragma unroll
            for (int v = 0; v < 8; ++v)
                out[((size_t)b * OO + (m0 + v)) * TT + (t + 1)] = acc[v];
        }
    }
}

__global__ void zero_col0(float* __restrict__ out) {
    int tid = blockIdx.x * blockDim.x + threadIdx.x;
    if (tid < BB * OO) out[(size_t)tid * TT] = 0.f;
}

// ---------------------------------------------------------------- launcher
extern "C" void kernel_launch(void* const* d_in, const int* in_sizes, int n_in,
                              void* d_out, int out_size, void* d_ws, size_t ws_size,
                              hipStream_t stream) {
    const float* features   = (const float*)d_in[0];
    const int*   fwd        = (const int*)d_in[1];
    const float* embed      = (const float*)d_in[2];
    const float* up_w       = (const float*)d_in[3];
    const float* up_b       = (const float*)d_in[4];
    const float* cond_w     = (const float*)d_in[5];
    const float* cond_b     = (const float*)d_in[6];
    const float* dil_w      = (const float*)d_in[7];
    const float* dil_b      = (const float*)d_in[8];
    const float* res_w      = (const float*)d_in[9];
    const float* res_b      = (const float*)d_in[10];
    const float* skip_w     = (const float*)d_in[11];
    const float* skip_b     = (const float*)d_in[12];
    const float* conv_out_w = (const float*)d_in[13];
    const float* conv_end_w = (const float*)d_in[14];
    float* out = (float*)d_out;

    char* ws = (char*)d_ws;
    size_t off = 0;
    auto alloc = [&](size_t bytes) -> char* {
        char* p = ws + off;
        off = (off + bytes + 255) & ~(size_t)255;
        return p;
    };
    bf16*  Wcat   = (bf16*)alloc((size_t)LL * MGATE * KCAT * 2);
    float* biasin = (float*)alloc((size_t)LL * MGATE * 4);
    bf16*  RS     = (bf16*)alloc((size_t)LL * MRS * RR * 2);
    bf16*  Wco    = (bf16*)alloc((size_t)OO * SS * 2);
    bf16*  Wce    = (bf16*)alloc((size_t)OO * OO * 2);
    bf16*  cond   = (bf16*)alloc((size_t)BB * TT * CCP * 2);
    float* xf[2];
    bf16*  xb[2];
    xf[0] = (float*)alloc((size_t)BB * TT * RR * 4);
    xf[1] = (float*)alloc((size_t)BB * TT * RR * 4);
    xb[0] = (bf16*)alloc((size_t)BB * TT * RR * 2);
    xb[1] = (bf16*)alloc((size_t)BB * TT * RR * 2);
    float* skipacc = (float*)alloc((size_t)BB * TT * SS * 4);
    bf16*  y1      = (bf16*)alloc((size_t)BB * TT * OO * 2);

    prep_weights<<<(LL * MGATE * KCAT + LL * MRS * RR + 2 * OO * SS + LL * MGATE + 255) / 256,
                   256, 0, stream>>>(dil_w, cond_w, res_w, skip_w, conv_out_w,
                                     conv_end_w, dil_b, cond_b,
                                     Wcat, RS, Wco, Wce, biasin);
    cond_kernel<<<(BB * TT * CCP + 255) / 256, 256, 0, stream>>>(features, up_w, up_b, cond);
    embed_kernel<<<(BB * TT * RR + 255) / 256, 256, 0, stream>>>(fwd, embed, xf[0], xb[0]);

    dim3 grid((TT + 127) / 128, BB);
    for (int i = 0; i < LL; ++i) {
        int d   = 1 << (i & 7);
        int in  = i & 1;
        int on  = in ^ 1;
        int rbi = (i < LL - 1) ? i : (LL - 2);
        layer_kernel<<<grid, 256, 0, stream>>>(
            Wcat + (size_t)i * MGATE * KCAT, biasin + (size_t)i * MGATE,
            RS + (size_t)i * MRS * RR, res_b + (size_t)rbi * RR,
            skip_b + (size_t)i * SS,
            xb[in], xf[in], cond, xb[on], xf[on], skipacc,
            d, (i == 0) ? 1 : 0, (i == LL - 1) ? 1 : 0);
    }
    conv_out_kernel<<<grid, 256, 0, stream>>>(Wco, skipacc, y1);
    conv_end_kernel<<<grid, 256, 0, stream>>>(Wce, y1, out);
    zero_col0<<<(BB * OO + 255) / 256, 256, 0, stream>>>(out);
}
